// MambaBlock_1133871366245
// MI455X (gfx1250) — matrix operand order repaired and checked by reading, not compiled, not run on hardware
//
#include <hip/hip_runtime.h>
#include <hip/hip_bf16.h>

// Problem constants (match reference)
#define HDIM   1024
#define BSZ    4
#define LSEQ   2048
#define MROWS  (BSZ * LSEQ)        // 8192
#define NCHUNK 16                  // L-chunks for blocked scan / reductions
#define LCHUNK (LSEQ / NCHUNK)     // 128

typedef __attribute__((ext_vector_type(16))) __bf16 v16bf;
typedef __attribute__((ext_vector_type(8)))  float  v8f;
typedef __attribute__((ext_vector_type(4)))  unsigned int v4u;
typedef __attribute__((ext_vector_type(8)))  int v8i;
typedef __attribute__((ext_vector_type(4)))  int v4i;

union Frag {
    v16bf v;
    uint4 q[2];
};

// Use the Tensor Data Mover when the toolchain exposes it.
#if defined(__gfx1250__) && __has_builtin(__builtin_amdgcn_tensor_load_to_lds) && \
    __has_builtin(__builtin_amdgcn_s_wait_tensorcnt)
#define USE_TDM 1
#else
#define USE_TDM 0
#endif

#if USE_TDM
// Issue a 2D TDM load: nrows x 32 bf16 tile, row stride HDIM elements,
// into LDS at lds_addr (contiguous [row][32] layout).  D# per ISA §8.3/8.4.
__device__ __forceinline__ void tdm_load_2d(unsigned lds_addr,
                                            const void* gaddr, int nrows) {
    const unsigned long long ga = (unsigned long long)(uintptr_t)gaddr;
    const v4u g0 = { 1u,                                   // count=1, user desc
                     lds_addr,                             // lds_addr[31:0]
                     (unsigned)(ga & 0xFFFFFFFFu),         // global_addr[31:0]
                     (unsigned)(((ga >> 32) & 0x01FFFFFFu) // global_addr[56:32]
                                | 0x80000000u) };          // type=2 (image)
    const v8i g1 = { 0x00010000,        // wg_mask=0, data_size=1 (2 bytes)
                     32 << 16,          // tensor_dim0 = 32   (bits 63:48)
                     nrows << 16,       // tensor_dim1 = rows (bits 95:80)
                     32 << 16,          // tile_dim0   = 32   (bits 127:112)
                     nrows,             // tile_dim1   = rows (bits 143:128)
                     HDIM,              // tensor_dim0_stride (bits 191:160)
                     0, 0 };
    const v4i gz = { 0, 0, 0, 0 };
#if __clang_major__ >= 23
    const v8i gz8 = { 0, 0, 0, 0, 0, 0, 0, 0 };
    __builtin_amdgcn_tensor_load_to_lds(g0, g1, gz, gz, gz8, 0);
#else
    __builtin_amdgcn_tensor_load_to_lds(g0, g1, gz, gz, 0);
#endif
}
#endif

// ---------------------------------------------------------------------------
// Shared bf16 WMMA mainloop: 128x256 block tile, 64x64 per wave, as a 4x4
// grid of v_wmma_f32_16x16x32_bf16 (K-step = 32).  Tiles are staged in LDS
// by the Tensor Data Mover, double-buffered so the next tile's DMA overlaps
// the current tile's WMMAs (TENSORcnt in-order => wait <=2 frees buffer i).
// ---------------------------------------------------------------------------
__device__ __forceinline__ void gemm_mainloop(const __bf16* __restrict__ Abase,
                                              const __bf16* __restrict__ Wtile,
                                              v8f (&acc)[4][4]) {
    __shared__ __bf16 sA[2][128 * 32];   // [buf][m][k]  2 x  8 KB
    __shared__ __bf16 sB[2][256 * 32];   // [buf][n][k]  2 x 16 KB

    const int t    = threadIdx.x;          // 0..255 (8 waves)
    const int lane = t & 31;
    const int wm   = ((t >> 5) & 1) * 64;  // wave M offset in block tile
    const int wn   = (t >> 6) * 64;        // wave N offset in block tile
    const int lr   = lane & 15;
    const int hi   = (lane < 16) ? 0 : 1;  // half-wave select

    const v8f vzero = {0.f, 0.f, 0.f, 0.f, 0.f, 0.f, 0.f, 0.f};
#pragma unroll
    for (int i = 0; i < 4; ++i)
#pragma unroll
        for (int j = 0; j < 4; ++j) acc[i][j] = vzero;

#if USE_TDM
    const unsigned ldsA[2] = { (unsigned)(uintptr_t)&sA[0][0],
                               (unsigned)(uintptr_t)&sA[1][0] };
    const unsigned ldsB[2] = { (unsigned)(uintptr_t)&sB[0][0],
                               (unsigned)(uintptr_t)&sB[1][0] };
    if (t < 32) {                        // prologue: fill buffer 0
        tdm_load_2d(ldsA[0], Abase, 128);
        tdm_load_2d(ldsB[0], Wtile, 256);
    }
#else
    const int sAm = t >> 1;              // 0..127 : A row
    const int sAk = (t & 1) * 16;        // 0/16   : A k-half
#endif

    int buf = 0;
    for (int kk = 0; kk < HDIM; kk += 32) {
        __syncthreads();   // consumers of buf^1 (iter kk-32) are done
#if USE_TDM
        if (t < 32) {      // wave 0 drives the Tensor Data Mover
            if (kk + 32 < HDIM) {
                tdm_load_2d(ldsA[buf ^ 1], Abase + kk + 32, 128);
                tdm_load_2d(ldsB[buf ^ 1], Wtile + kk + 32, 256);
                // tensor ops complete in order: <=2 outstanding means the
                // two loads for the *current* buffer have landed.
                __builtin_amdgcn_s_wait_tensorcnt(2);
            } else {
                __builtin_amdgcn_s_wait_tensorcnt(0);
            }
        }
#else
        // Manual staging fallback: load all temporaries first so the
        // global_load_b128s issue back-to-back.
        const __bf16* ga = Abase + (size_t)sAm * HDIM + kk + sAk;
        const __bf16* gb = Wtile + (size_t)t * HDIM + kk;
        const uint4 ta0 = *(const uint4*)(ga);
        const uint4 ta1 = *(const uint4*)(ga + 8);
        const uint4 tb0 = *(const uint4*)(gb);
        const uint4 tb1 = *(const uint4*)(gb + 8);
        const uint4 tb2 = *(const uint4*)(gb + 16);
        const uint4 tb3 = *(const uint4*)(gb + 24);
        *(uint4*)&sA[buf][sAm * 32 + sAk]     = ta0;
        *(uint4*)&sA[buf][sAm * 32 + sAk + 8] = ta1;
        *(uint4*)&sB[buf][t * 32 +  0] = tb0;
        *(uint4*)&sB[buf][t * 32 +  8] = tb1;
        *(uint4*)&sB[buf][t * 32 + 16] = tb2;
        *(uint4*)&sB[buf][t * 32 + 24] = tb3;
        if (kk + 32 < HDIM) {            // global_prefetch_b8 next K-tile
            __builtin_prefetch(ga + 32, 0, 3);
            __builtin_prefetch(gb + 32, 0, 3);
        }
#endif
        __syncthreads();

        // A fragments: lane holds row M=lr; lanes<16: K {0..7,16..23},
        // lanes>=16: K {8..15,24..31}  (ISA 7.12.2, 16-bit A 16x32)
        Frag a[4], b[4];
        const __bf16* cA = &sA[buf][0];
        const __bf16* cB = &sB[buf][0];
        const int kA = hi * 8;
#pragma unroll
        for (int i = 0; i < 4; ++i) {
            const int row = wm + i * 16 + lr;
            a[i].q[0] = *(const uint4*)&cA[row * 32 + kA];
            a[i].q[1] = *(const uint4*)&cA[row * 32 + kA + 16];
        }
        // B fragments: lane holds col N=lr; lanes<16: K {0..7,16..23}, lanes>=16: K {8..15,24..31}
        const int kB = hi * 8;
#pragma unroll
        for (int j = 0; j < 4; ++j) {
            const int col = wn + j * 16 + lr;
            b[j].q[0] = *(const uint4*)&cB[col * 32 + kB];
            b[j].q[1] = *(const uint4*)&cB[col * 32 + kB + 16];
        }
#pragma unroll
        for (int i = 0; i < 4; ++i)
#pragma unroll
            for (int j = 0; j < 4; ++j)
                acc[i][j] = __builtin_amdgcn_wmma_f32_16x16x32_bf16(
                    false, a[i].v, false, b[j].v, (short)0, acc[i][j],
                    false, false);
        buf ^= 1;
    }
    __syncthreads();
}

// ---------------------------------------------------------------------------
// GEMM1: proj = x @ W_proj.T, skipping unused Bs columns.
// Logical col n in [0,3072): n<1024 -> delta_raw (bf16), 1024..2047 -> Cs,
// 2048..3071 -> residual (both f32 in P at col n-1024).  Weight row = n
// (n<1024) else n+1024.
// ---------------------------------------------------------------------------
__global__ void __launch_bounds__(256)
gemm_proj_kernel(const __bf16* __restrict__ xb, const __bf16* __restrict__ wpb,
                 __bf16* __restrict__ drb, float* __restrict__ P) {
    v8f acc[4][4];
    const int mBlock = blockIdx.x * 128;
    const int nBlock = blockIdx.y * 256;                       // 256-aligned
    const int wBase  = (nBlock < 1024) ? nBlock : nBlock + 1024;
    gemm_mainloop(xb + (size_t)mBlock * HDIM, wpb + (size_t)wBase * HDIM, acc);

    const int t = threadIdx.x, lane = t & 31;
    const int wm = ((t >> 5) & 1) * 64, wn = (t >> 6) * 64;
    const int rOff = (lane >> 4) * 8, cOff = lane & 15;        // C/D layout
#pragma unroll
    for (int i = 0; i < 4; ++i)
#pragma unroll
        for (int j = 0; j < 4; ++j)
#pragma unroll
            for (int r = 0; r < 8; ++r) {
                const int row = mBlock + wm + i * 16 + rOff + r;
                const int col = nBlock + wn + j * 16 + cOff;
                const float v = acc[i][j][r];
                if (col < 1024)
                    drb[(size_t)row * 1024 + col] = (__bf16)v;
                else
                    P[(size_t)row * 2048 + (col - 1024)] = v;
            }
}

// ---------------------------------------------------------------------------
// GEMM2: delta = softplus(delta_raw @ W_delta.T + b_delta)
// ---------------------------------------------------------------------------
__global__ void __launch_bounds__(256)
gemm_delta_kernel(const __bf16* __restrict__ drb, const __bf16* __restrict__ wdb,
                  const float* __restrict__ bias, float* __restrict__ delta) {
    v8f acc[4][4];
    const int mBlock = blockIdx.x * 128;
    const int nBlock = blockIdx.y * 256;
    gemm_mainloop(drb + (size_t)mBlock * HDIM, wdb + (size_t)nBlock * HDIM, acc);

    const int t = threadIdx.x, lane = t & 31;
    const int wm = ((t >> 5) & 1) * 64, wn = (t >> 6) * 64;
    const int rOff = (lane >> 4) * 8, cOff = lane & 15;
#pragma unroll
    for (int i = 0; i < 4; ++i)
#pragma unroll
        for (int j = 0; j < 4; ++j)
#pragma unroll
            for (int r = 0; r < 8; ++r) {
                const int row = mBlock + wm + i * 16 + rOff + r;
                const int col = nBlock + wn + j * 16 + cOff;
                float v = acc[i][j][r] + bias[col];
                v = (v > 20.0f) ? v : log1pf(expf(v));   // softplus
                delta[(size_t)row * 1024 + col] = v;
            }
}

// ---------------------------------------------------------------------------
// Elementwise / reduction kernels
// ---------------------------------------------------------------------------
__global__ void cvt_bf16_kernel(const float* __restrict__ s, __bf16* __restrict__ d,
                                int n) {
    const int stride = gridDim.x * blockDim.x;
    for (int i = blockIdx.x * blockDim.x + threadIdx.x; i < n; i += stride)
        d[i] = (__bf16)s[i];
}

__global__ void zero_f32_kernel(float* __restrict__ p, int n) {
    const int i = blockIdx.x * blockDim.x + threadIdx.x;
    if (i < n) p[i] = 0.0f;
}

// uC[b,h] = sum_l x*Cs ; Csum[b,h] = sum_l Cs   (atomic over L-chunks)
__global__ void __launch_bounds__(1024)
reduce_uc_kernel(const float* __restrict__ x, const float* __restrict__ P,
                 float* __restrict__ uC, float* __restrict__ Cs) {
    const int b = blockIdx.x >> 4;
    const int c = blockIdx.x & 15;
    const int h = threadIdx.x;
    float su = 0.f, sc = 0.f;
    for (int l = c * LCHUNK; l < (c + 1) * LCHUNK; ++l) {
        const size_t rl = (size_t)b * LSEQ + l;
        const float cv = P[rl * 2048 + h];        // Cs column
        su += x[rl * HDIM + h] * cv;
        sc += cv;
    }
    atomicAdd(&uC[b * HDIM + h], su);
    atomicAdd(&Cs[b * HDIM + h], sc);
}

// chunkSums[b,c,h] = B1[h] * sum_{l in chunk c} delta[b,l,h]
__global__ void __launch_bounds__(1024)
scan_partial_kernel(const float* __restrict__ delta, const float* __restrict__ Bp,
                    float* __restrict__ csums) {
    const int b = blockIdx.x >> 4;
    const int c = blockIdx.x & 15;
    const int h = threadIdx.x;
    float s = 0.f;
    for (int l = c * LCHUNK; l < (c + 1) * LCHUNK; ++l)
        s += delta[((size_t)b * LSEQ + l) * HDIM + h];
    csums[((size_t)b * NCHUNK + c) * HDIM + h] = s * Bp[h];
}

// out = exp(delta*A1)*B1*uC + cumsum(delta*B1)*Csum + residual*D
__global__ void __launch_bounds__(1024)
final_kernel(const float* __restrict__ delta, const float* __restrict__ P,
             const float* __restrict__ A, const float* __restrict__ Bp,
             const float* __restrict__ Dv, const float* __restrict__ uC,
             const float* __restrict__ Cs, const float* __restrict__ csums,
             float* __restrict__ out) {
    const int b = blockIdx.x >> 4;
    const int c = blockIdx.x & 15;
    const int h = threadIdx.x;
    const float a1 = A[h], b1 = Bp[h], d0 = Dv[0];
    const float uc = uC[b * HDIM + h], cs = Cs[b * HDIM + h];
    float run = 0.f;                                   // exclusive prefix of chunks
    for (int cc = 0; cc < c; ++cc)
        run += csums[((size_t)b * NCHUNK + cc) * HDIM + h];
    for (int l = c * LCHUNK; l < (c + 1) * LCHUNK; ++l) {
        const size_t rl = (size_t)b * LSEQ + l;
        const float d = delta[rl * HDIM + h];
        run += d * b1;                                 // inclusive cumsum
        const float dA  = expf(d * a1) * b1;
        const float res = P[rl * 2048 + 1024 + h];     // residual column
        out[rl * HDIM + h] = dA * uc + run * cs + res * d0;
    }
}

// ---------------------------------------------------------------------------
// Launch
// ---------------------------------------------------------------------------
extern "C" void kernel_launch(void* const* d_in, const int* in_sizes, int n_in,
                              void* d_out, int out_size, void* d_ws, size_t ws_size,
                              hipStream_t stream) {
    const float* x   = (const float*)d_in[0];   // (B,L,H)
    const float* Wp  = (const float*)d_in[1];   // (4H,H)
    const float* A   = (const float*)d_in[2];   // (H)
    const float* Bp  = (const float*)d_in[3];   // (H)
    const float* Dv  = (const float*)d_in[4];   // (1)
    const float* Wd  = (const float*)d_in[5];   // (H,H)
    const float* bd  = (const float*)d_in[6];   // (H)
    float* out = (float*)d_out;

    char* w = (char*)d_ws;
    __bf16* xb  = (__bf16*)w;  w += (size_t)MROWS * HDIM * 2;      // 16 MB
    __bf16* wpb = (__bf16*)w;  w += (size_t)4 * HDIM * HDIM * 2;   //  8 MB
    __bf16* wdb = (__bf16*)w;  w += (size_t)HDIM * HDIM * 2;       //  2 MB
    __bf16* drb = (__bf16*)w;  w += (size_t)MROWS * HDIM * 2;      // 16 MB
    float* P    = (float*)w;   w += (size_t)MROWS * 2048 * 4;      // 64 MB (Cs|res)
    float* delta= (float*)w;   w += (size_t)MROWS * HDIM * 4;      // 32 MB
    float* uC   = (float*)w;   w += (size_t)BSZ * HDIM * 4;
    float* Csum = (float*)w;   w += (size_t)BSZ * HDIM * 4;
    float* csums= (float*)w;   w += (size_t)BSZ * NCHUNK * HDIM * 4;

    // 1) f32 -> bf16 conversions
    cvt_bf16_kernel<<<4096, 256, 0, stream>>>(x,  xb,  MROWS * HDIM);
    cvt_bf16_kernel<<<4096, 256, 0, stream>>>(Wp, wpb, 4 * HDIM * HDIM);
    cvt_bf16_kernel<<<1024, 256, 0, stream>>>(Wd, wdb, HDIM * HDIM);
    // 2) zero uC | Csum (contiguous)
    zero_f32_kernel<<<(2 * BSZ * HDIM + 255) / 256, 256, 0, stream>>>(uC, 2 * BSZ * HDIM);
    // 3) GEMM1: M=8192, N=3072(eff), K=1024
    gemm_proj_kernel<<<dim3(MROWS / 128, 3072 / 256), 256, 0, stream>>>(xb, wpb, drb, P);
    // 4) GEMM2 + bias + softplus: M=8192, N=1024, K=1024
    gemm_delta_kernel<<<dim3(MROWS / 128, HDIM / 256), 256, 0, stream>>>(drb, wdb, bd, delta);
    // 5) uC / Csum reductions over L
    reduce_uc_kernel<<<BSZ * NCHUNK, 1024, 0, stream>>>(x, P, uC, Csum);
    // 6) blocked scan: chunk partial sums, then fused prefix + output
    scan_partial_kernel<<<BSZ * NCHUNK, 1024, 0, stream>>>(delta, Bp, csums);
    final_kernel<<<BSZ * NCHUNK, 1024, 0, stream>>>(delta, P, A, Bp, Dv, uC, Csum,
                                                    csums, out);
}
